// Basenet_50345606644034
// MI455X (gfx1250) — compile-verified, matrix operand order checked by reading, not run on hardware
//
#include <hip/hip_runtime.h>
#include <cstddef>

#define D5 512
#define G4 2048
#define TLEN 16
#define FLEN 50
#define CAND 10
#define BSZ 16
#define NFACT 8000   /* BS*CAND*FLEN */
#define B2 160       /* BS*CAND */
#define LDIM 128
#define NEGV -9999999.f
#define EPSV 1e-5f

typedef __attribute__((ext_vector_type(16))) __bf16 v16bf;
typedef __attribute__((ext_vector_type(8)))  float  v8f;

__device__ __forceinline__ unsigned short f2bf(float f) {
    unsigned u = __float_as_uint(f);
    u += 0x7FFFu + ((u >> 16) & 1u);   // round-to-nearest-even
    return (unsigned short)(u >> 16);
}
__device__ __forceinline__ float sigm(float x) { return 1.f / (1.f + expf(-x)); }

// ---------------------------------------------------------------------------
// bf16 WMMA GEMM:  C[M x N] = act((A * B^T + bias) * scale + shift)
// A: M x K bf16 row-major (lda). B: N x K bf16 row-major (weight layout, ldb).
// Block = 4 waves; block tile = 128(M) x 64(N); each wave owns 2 M-tiles.
// B tile (64x32) is staged in LDS via async-to-LDS loads, double-buffered so
// the next K-step's fill overlaps the current K-step's 8 WMMAs.
// Requires: M%16==0, N%64==0, K%32==0.
// ---------------------------------------------------------------------------
__global__ __launch_bounds__(128) void gemm_bf16_wmma(
    const unsigned short* __restrict__ A, const unsigned short* __restrict__ B,
    const float* __restrict__ bias, const float* __restrict__ scale,
    const float* __restrict__ shift, float* __restrict__ C,
    int M, int N, int K, int lda, int ldb, int ldc, int act)
{
    __shared__ unsigned short smem[2][64 * 32];   // 2 x 4KB B tiles
    const int tid  = threadIdx.x;
    const int lane = tid & 31;
    const int wave = tid >> 5;
    const int half = lane >> 4;
    const int row  = lane & 15;     // A M-row / B N-col within a 16-tile
    const int n0 = blockIdx.x * 64;
    const int mb = blockIdx.y * 128 + wave * 32;  // this wave's 2 M-tiles
    const bool v0 = (mb < M);
    const bool v1 = (mb + 16 < M);
    const int r0 = v0 ? (mb + row) : row;         // clamped A rows
    const int r1 = v1 ? (mb + 16 + row) : row;

    // Issue one 64x32 B-tile fill into smem[p] for K-offset k0.
    // 256 chunks of 16B; each thread issues 2 async b128 loads (2/wave inst).
    auto fill = [&](int p, int k0) {
        #pragma unroll
        for (int i = 0; i < 2; ++i) {
            int ch = tid + i * 128;               // 0..255
            int r  = ch >> 2;                     // B row (N dim)
            int pc = ch & 3;                      // 16B piece within row
            const unsigned short* g = B + (size_t)(n0 + r) * ldb + k0 + pc * 8;
            unsigned laddr = (unsigned)(size_t)(&smem[p][ch * 8]);
            asm volatile("global_load_async_to_lds_b128 %0, %1, off"
                         :: "v"(laddr), "v"((unsigned long long)(size_t)g)
                         : "memory");
        }
    };

    v8f acc[2][4] = {};
    fill(0, 0);
    for (int k0 = 0; k0 < K; k0 += 32) {
        const int p = (k0 >> 5) & 1;
        // A tiles from global (per wave)
        union { unsigned u[8]; v16bf v; } a0, a1;
        #pragma unroll
        for (int r = 0; r < 8; ++r) {
            int koff = ((r < 4) ? (2 * r) : (16 + 2 * (r - 4))) + 8 * half;
            a0.u[r] = *(const unsigned*)(A + (size_t)r0 * lda + k0 + koff);
            a1.u[r] = *(const unsigned*)(A + (size_t)r1 * lda + k0 + koff);
        }
        if (k0 + 32 < K) {
            fill(p ^ 1, k0 + 32);                 // prefetch next tile
            asm volatile("s_wait_asynccnt 0x2" ::: "memory");  // current tile done
        } else {
            asm volatile("s_wait_asynccnt 0x0" ::: "memory");  // drain
        }
        __syncthreads();                          // fill(p) visible to all waves
        #pragma unroll
        for (int j = 0; j < 4; ++j) {
            union { unsigned u[8]; v16bf v; } b;
            const unsigned short* bp = &smem[p][(j * 16 + row) * 32];
            #pragma unroll
            for (int r = 0; r < 8; ++r) {
                int koff = ((r < 4) ? (2 * r) : (16 + 2 * (r - 4))) + 8 * half;
                b.u[r] = *(const unsigned*)(bp + koff);
            }
            acc[0][j] = __builtin_amdgcn_wmma_f32_16x16x32_bf16(
                false, a0.v, false, b.v, (short)0, acc[0][j], false, false);
            acc[1][j] = __builtin_amdgcn_wmma_f32_16x16x32_bf16(
                false, a1.v, false, b.v, (short)0, acc[1][j], false, false);
        }
        __syncthreads();                          // done reading p before refill
    }
    #pragma unroll
    for (int t = 0; t < 2; ++t) {
        if (!(t == 0 ? v0 : v1)) continue;
        int mt = mb + t * 16;
        #pragma unroll
        for (int j = 0; j < 4; ++j) {
            int n = n0 + j * 16 + row;
            float bs = bias  ? bias[n]  : 0.f;
            float sc = scale ? scale[n] : 1.f;
            float sh = scale ? shift[n] : 0.f;
            #pragma unroll
            for (int v = 0; v < 8; ++v) {
                float val = acc[t][j][v] + bs;
                val = val * sc + sh;
                if (act == 1) val = tanhf(val);
                else if (act == 2) val = fmaxf(val, 0.f);
                C[(size_t)(mt + v + 8 * half) * ldc + n] = val;
            }
        }
    }
}

// --------------------------- helper kernels --------------------------------
__global__ void k_cvt_bf16(const float* __restrict__ s, unsigned short* __restrict__ d, int n) {
    int i = blockIdx.x * blockDim.x + threadIdx.x;
    if (i < n) d[i] = f2bf(s[i]);
}

// dst[n, 0:Ka] = Wa[n,:], dst[n, Ka:Ka+Kb] = Wb[n,:]  (bf16)
__global__ void k_cat2_bf16(const float* __restrict__ Wa, const float* __restrict__ Wb,
                            unsigned short* __restrict__ dst, int N, int Ka, int Kb) {
    int Kt = Ka + Kb;
    int i = blockIdx.x * blockDim.x + threadIdx.x;
    if (i >= N * Kt) return;
    int nrow = i / Kt, k = i % Kt;
    float v = (k < Ka) ? Wa[(size_t)nrow * Ka + k] : Wb[(size_t)nrow * Kb + (k - Ka)];
    dst[i] = f2bf(v);
}

// Embedding gather: one block per row; copies 512 bf16 (=256 dwords).
__global__ void k_embed(const int* __restrict__ ids, int id_stride, int id_off,
                        const unsigned short* __restrict__ emb, unsigned short* __restrict__ dst,
                        int ld_dst) {
    int r = blockIdx.x;
    int id = ids[(size_t)r * id_stride + id_off];
    const unsigned* s = (const unsigned*)(emb + (size_t)id * D5);
    unsigned* d = (unsigned*)(dst + (size_t)r * ld_dst);
    for (int i = threadIdx.x; i < D5 / 2; i += blockDim.x) d[i] = s[i];
}

// LSTM cell (seq form): update c, write h (bf16) into xh[:,512:1024],
// accumulate acc += w_row * h   (w_row = accw[r*accw_stride + accw_off]).
__global__ void k_cell_seq(const float* __restrict__ gates, const float* __restrict__ b,
                           float* __restrict__ c, unsigned short* __restrict__ xh,
                           float* __restrict__ acc, const float* __restrict__ accw,
                           int accw_stride, int accw_off, int Mrows) {
    int idx = blockIdx.x * blockDim.x + threadIdx.x;
    if (idx >= Mrows * D5) return;
    int r = idx / D5, d = idx % D5;
    const float* g = gates + (size_t)r * G4;
    float i_ = sigm(g[d] + b[d]);
    float f_ = sigm(g[D5 + d] + b[D5 + d]);
    float g_ = tanhf(g[2 * D5 + d] + b[2 * D5 + d]);
    float o_ = sigm(g[3 * D5 + d] + b[3 * D5 + d]);
    float cn = f_ * c[idx] + i_ * g_;
    float hn = o_ * tanhf(cn);
    c[idx] = cn;
    xh[(size_t)r * 1024 + D5 + d] = f2bf(hn);
    acc[idx] += accw[(size_t)r * accw_stride + accw_off] * hn;
}

// Episodic attention cell: gated h_pre update, accumulate eh/ec with ffm.
__global__ void k_cell_epi(const float* __restrict__ gates, const float* __restrict__ b,
                           float* __restrict__ c, float* __restrict__ hpre,
                           unsigned short* __restrict__ xh, const float* __restrict__ w, int s,
                           const float* __restrict__ ffm, float* __restrict__ eh,
                           float* __restrict__ ec) {
    int idx = blockIdx.x * blockDim.x + threadIdx.x;
    if (idx >= B2 * D5) return;
    int r = idx / D5, d = idx % D5;
    const float* g = gates + (size_t)r * G4;
    float i_ = sigm(g[d] + b[d]);
    float f_ = sigm(g[D5 + d] + b[D5 + d]);
    float g_ = tanhf(g[2 * D5 + d] + b[2 * D5 + d]);
    float o_ = sigm(g[3 * D5 + d] + b[3 * D5 + d]);
    float cn = f_ * c[idx] + i_ * g_;
    float hn = o_ * tanhf(cn);
    float wt = w[r * FLEN + s];
    float hp = wt * hn + (1.f - wt) * hpre[idx];
    c[idx] = cn; hpre[idx] = hp;
    xh[(size_t)r * 1024 + D5 + d] = f2bf(hp);
    float fw = ffm[r * FLEN + s];
    eh[idx] += fw * hp;
    ec[idx] += fw * cn;
}

// Memory-update cell: m = sig(o)*tanh(sig(f)*ec + sig(i)*tanh(g))
__global__ void k_cell_mem(const float* __restrict__ gates, const float* __restrict__ b,
                           const float* __restrict__ ec, float* __restrict__ m) {
    int idx = blockIdx.x * blockDim.x + threadIdx.x;
    if (idx >= B2 * D5) return;
    int r = idx / D5, d = idx % D5;
    const float* g = gates + (size_t)r * G4;
    float i_ = sigm(g[d] + b[d]);
    float f_ = sigm(g[D5 + d] + b[D5 + d]);
    float g_ = tanhf(g[2 * D5 + d] + b[2 * D5 + d]);
    float o_ = sigm(g[3 * D5 + d] + b[3 * D5 + d]);
    float cp = f_ * ec[idx] + i_ * g_;
    m[idx] = o_ * tanhf(cp);
}

// BatchNorm over rows x cols (in-place safe); optional bf16 copy.
__global__ void k_bn(const float* __restrict__ x, float* __restrict__ y,
                     unsigned short* __restrict__ ybf, const float* __restrict__ g,
                     const float* __restrict__ bt, const float* __restrict__ mu,
                     const float* __restrict__ var, int rows, int cols) {
    int idx = blockIdx.x * blockDim.x + threadIdx.x;
    if (idx >= rows * cols) return;
    int d = idx % cols;
    float v = g[d] * (x[idx] - mu[d]) * rsqrtf(var[d] + EPSV) + bt[d];
    y[idx] = v;
    if (ybf) ybf[idx] = f2bf(v);
}

__global__ void k_bn_eff(const float* __restrict__ g, const float* __restrict__ bt,
                         const float* __restrict__ mu, const float* __restrict__ var,
                         float* __restrict__ scale, float* __restrict__ shift, int n) {
    int i = blockIdx.x * blockDim.x + threadIdx.x;
    if (i >= n) return;
    float s = g[i] * rsqrtf(var[i] + EPSV);
    scale[i] = s;
    shift[i] = bt[i] - s * mu[i];
}

// Global attention: one block per batch, 512 threads.
__global__ void k_att_glob(const float* __restrict__ gg, const float* __restrict__ gq,
                           const float* __restrict__ gfcW, const float* __restrict__ gfcb,
                           const float* __restrict__ feat, float* __restrict__ fglob) {
    int b = blockIdx.x, tid = threadIdx.x;
    __shared__ float red[512];
    __shared__ float wn[49];
    float gqd = gq[b * D5 + tid];
    float gw = gfcW[tid];
    for (int n = 0; n < 49; ++n) {
        red[tid] = tanhf(gg[((size_t)b * 49 + n) * D5 + tid] + gqd) * gw;
        __syncthreads();
        for (int off = 256; off > 0; off >>= 1) {
            if (tid < off) red[tid] += red[tid + off];
            __syncthreads();
        }
        if (tid == 0) wn[n] = red[0] + gfcb[0];
        __syncthreads();
    }
    if (tid == 0) {
        float mx = -1e30f;
        for (int n = 0; n < 49; ++n) mx = fmaxf(mx, wn[n]);
        float s = 0.f;
        for (int n = 0; n < 49; ++n) { wn[n] = expf(wn[n] - mx); s += wn[n]; }
        float inv = 1.f / s;
        for (int n = 0; n < 49; ++n) wn[n] *= inv;
    }
    __syncthreads();
    float a = 0.f;
    for (int n = 0; n < 49; ++n) a += wn[n] * feat[((size_t)b * 49 + n) * D5 + tid];
    fglob[b * D5 + tid] = a;
}

// Position FC (5->128) + BN + relu: one block per row, 128 threads.
__global__ void k_pos(const float* __restrict__ loc, const float* __restrict__ W,
                      const float* __restrict__ b, const float* __restrict__ g,
                      const float* __restrict__ bt, const float* __restrict__ mu,
                      const float* __restrict__ var, float* __restrict__ out) {
    int r = blockIdx.x, o = threadIdx.x;
    float s = b[o];
    #pragma unroll
    for (int j = 0; j < 5; ++j) s += loc[r * 5 + j] * W[o * 5 + j];
    s = g[o] * (s - mu[o]) * rsqrtf(var[o] + EPSV) + bt[o];
    out[r * LDIM + o] = fmaxf(s, 0.f);
}

// Row dot with g2_W + fact-mask: wave per row.
__global__ void k_g2(const float* __restrict__ zh, const float* __restrict__ g2W,
                     const float* __restrict__ g2b, const int* __restrict__ fmask,
                     float* __restrict__ Z, int rows) {
    int wave = threadIdx.x >> 5, lane = threadIdx.x & 31;
    int r = blockIdx.x * (blockDim.x >> 5) + wave;
    if (r >= rows) return;
    float s = 0.f;
    for (int d = lane; d < D5; d += 32) s += zh[(size_t)r * D5 + d] * g2W[d];
    for (int off = 16; off; off >>= 1) s += __shfl_xor(s, off);
    if (lane == 0) Z[r] = (fmask[r] == 0) ? NEGV : (s + g2b[0]);
}

// Softmax over FLEN per row: wave per row.
__global__ void k_softmax50(const float* __restrict__ Z, float* __restrict__ w) {
    int r = blockIdx.x, lane = threadIdx.x;
    float mx = -1e30f;
    for (int i = lane; i < FLEN; i += 32) mx = fmaxf(mx, Z[r * FLEN + i]);
    for (int off = 16; off; off >>= 1) mx = fmaxf(mx, __shfl_xor(mx, off));
    float s = 0.f;
    for (int i = lane; i < FLEN; i += 32) s += expf(Z[r * FLEN + i] - mx);
    for (int off = 16; off; off >>= 1) s += __shfl_xor(s, off);
    float inv = 1.f / s;
    for (int i = lane; i < FLEN; i += 32) w[r * FLEN + i] = expf(Z[r * FLEN + i] - mx) * inv;
}

// Build z = [ff*fe, ff*mm, |ff-fe|, |ff-mm|] (bf16, 8000x2048).
__global__ void k_build_z(const float* __restrict__ ff, const float* __restrict__ fm,
                          const float* __restrict__ m, unsigned short* __restrict__ z) {
    int idx = blockIdx.x * blockDim.x + threadIdx.x;
    if (idx >= NFACT * D5) return;
    int r = idx / D5, d = idx % D5, bc = r / FLEN;
    float a = ff[idx], fe = fm[bc * D5 + d], mm = m[bc * D5 + d];
    size_t base = (size_t)r * G4;
    z[base + d]          = f2bf(a * fe);
    z[base + D5 + d]     = f2bf(a * mm);
    z[base + 2 * D5 + d] = f2bf(fabsf(a - fe));
    z[base + 3 * D5 + d] = f2bf(fabsf(a - mm));
}

// Gather fact row s into xh_ep[:, 0:512] (bf16 copy).
__global__ void k_gather_fact(const unsigned short* __restrict__ ffbf,
                              unsigned short* __restrict__ xh, int s) {
    int bc = blockIdx.x;
    const unsigned* src = (const unsigned*)(ffbf + (size_t)(bc * FLEN + s) * D5);
    unsigned* d = (unsigned*)(xh + (size_t)bc * 1024);
    for (int i = threadIdx.x; i < D5 / 2; i += blockDim.x) d[i] = src[i];
}

__global__ void k_bcast_fm(const float* __restrict__ fexp, float* __restrict__ fm,
                           float* __restrict__ m) {
    int idx = blockIdx.x * blockDim.x + threadIdx.x;
    if (idx >= B2 * D5) return;
    int r = idx / D5, d = idx % D5;
    float v = fexp[(r / CAND) * D5 + d];
    fm[idx] = v; m[idx] = v;
}

__global__ void k_pack_eh_m(const float* __restrict__ eh, const float* __restrict__ m,
                            unsigned short* __restrict__ xh) {
    int idx = blockIdx.x * blockDim.x + threadIdx.x;
    if (idx >= B2 * D5) return;
    int r = idx / D5, d = idx % D5;
    xh[(size_t)r * 1024 + d] = f2bf(eh[idx]);
    xh[(size_t)r * 1024 + D5 + d] = f2bf(m[idx]);
}

__global__ void k_pack_feat(const float* __restrict__ m, const float* __restrict__ fglob,
                            const float* __restrict__ flocal, const float* __restrict__ floc,
                            unsigned short* __restrict__ feat) {
    int r = blockIdx.x;
    for (int d = threadIdx.x; d < 3 * D5 + LDIM; d += blockDim.x) {
        float v;
        if (d < D5) v = m[r * D5 + d];
        else if (d < 2 * D5) v = fglob[(r / CAND) * D5 + (d - D5)];
        else if (d < 3 * D5) v = flocal[r * D5 + (d - 2 * D5)];
        else v = floc[r * LDIM + (d - 3 * D5)];
        feat[(size_t)r * (3 * D5 + LDIM) + d] = f2bf(v);
    }
}

// Final scores + candidate softmax: one block per batch, 10 waves.
__global__ void k_scores(const float* __restrict__ fexp, const float* __restrict__ f,
                         const int* __restrict__ cmask, float* __restrict__ out) {
    int b = blockIdx.x;
    int wave = threadIdx.x >> 5, lane = threadIdx.x & 31;
    __shared__ float sc[CAND];
    if (wave < CAND) {
        int bc = b * CAND + wave;
        float s = 0.f;
        for (int d = lane; d < D5; d += 32) s += fexp[b * D5 + d] * f[(size_t)bc * D5 + d];
        for (int off = 16; off; off >>= 1) s += __shfl_xor(s, off);
        if (lane == 0) sc[wave] = (cmask[bc] == 0) ? NEGV : s;
    }
    __syncthreads();
    if (threadIdx.x == 0) {
        float mx = -1e30f;
        for (int i = 0; i < CAND; ++i) mx = fmaxf(mx, sc[i]);
        float s = 0.f;
        for (int i = 0; i < CAND; ++i) { sc[i] = expf(sc[i] - mx); s += sc[i]; }
        float inv = 1.f / s;
        for (int i = 0; i < CAND; ++i) sc[i] *= inv;
    }
    __syncthreads();
    if (threadIdx.x < CAND) out[b * CAND + threadIdx.x] = sc[threadIdx.x];
}

// ---------------------------------------------------------------------------
extern "C" void kernel_launch(void* const* d_in, const int* in_sizes, int n_in,
                              void* d_out, int out_size, void* d_ws, size_t ws_size,
                              hipStream_t stream) {
    (void)in_sizes; (void)n_in; (void)out_size; (void)ws_size;
    // ---- inputs -----------------------------------------------------------
    const int*   expression    = (const int*)d_in[0];
    const float* e_mask        = (const float*)d_in[1];
    const float* f_global_feat = (const float*)d_in[2];
    const float* f_local_feat  = (const float*)d_in[3];
    const float* locations     = (const float*)d_in[4];
    const int*   facts         = (const int*)d_in[5];
    const float* maskp         = (const float*)d_in[6];
    const int*   f_mask        = (const int*)d_in[7];
    const float* ff_mask       = (const float*)d_in[8];
    const int*   c_mask        = (const int*)d_in[9];
    const float* emb       = (const float*)d_in[10];
    const float* exp_Wih   = (const float*)d_in[11];
    const float* exp_Whh   = (const float*)d_in[12];
    const float* exp_b     = (const float*)d_in[13];
    const float* fact_Wih  = (const float*)d_in[14];
    const float* fact_Whh  = (const float*)d_in[15];
    const float* fact_b    = (const float*)d_in[16];
    const float* exp_bn_g  = (const float*)d_in[17];
    const float* exp_bn_b  = (const float*)d_in[18];
    const float* exp_bn_m  = (const float*)d_in[19];
    const float* exp_bn_v  = (const float*)d_in[20];
    const float* fact_bn_g = (const float*)d_in[21];
    const float* fact_bn_b = (const float*)d_in[22];
    const float* fact_bn_m = (const float*)d_in[23];
    const float* fact_bn_v = (const float*)d_in[24];
    const float* gq_W  = (const float*)d_in[25];
    const float* gq_b  = (const float*)d_in[26];
    const float* gg_W  = (const float*)d_in[27];
    const float* gg_b  = (const float*)d_in[28];
    const float* gfc_W = (const float*)d_in[29];
    const float* gfc_b = (const float*)d_in[30];
    const float* loc_W = (const float*)d_in[31];
    const float* loc_b = (const float*)d_in[32];
    const float* locbn_g = (const float*)d_in[33];
    const float* locbn_b = (const float*)d_in[34];
    const float* locbn_m = (const float*)d_in[35];
    const float* locbn_v = (const float*)d_in[36];
    const float* pos_W   = (const float*)d_in[37];
    const float* pos_b   = (const float*)d_in[38];
    const float* posbn_g = (const float*)d_in[39];
    const float* posbn_b = (const float*)d_in[40];
    const float* posbn_m = (const float*)d_in[41];
    const float* posbn_v = (const float*)d_in[42];
    const float* g1_W = (const float*)d_in[43];
    const float* g1_b = (const float*)d_in[44];
    const float* g2_W = (const float*)d_in[45];
    const float* g2_b = (const float*)d_in[46];
    const float* mn_Wih = (const float*)d_in[47];
    const float* mn_Whh = (const float*)d_in[48];
    const float* mn_b   = (const float*)d_in[49];
    const float* mc_Wih = (const float*)d_in[50];
    const float* mc_Whh = (const float*)d_in[51];
    const float* mc_b   = (const float*)d_in[52];
    const float* final_W = (const float*)d_in[53];
    const float* final_b = (const float*)d_in[54];
    float* out = (float*)d_out;

    // ---- workspace bump allocator ----------------------------------------
    char* base = (char*)d_ws;
    size_t off = 0;
    auto alloc = [&](size_t bytes) -> void* {
        void* p = base + off;
        off = (off + bytes + 255) & ~(size_t)255;
        return p;
    };
    unsigned short* emb_bf    = (unsigned short*)alloc((size_t)10000 * D5 * 2);
    unsigned short* wcat_exp  = (unsigned short*)alloc((size_t)G4 * 1024 * 2);
    unsigned short* wcat_fact = (unsigned short*)alloc((size_t)G4 * 1024 * 2);
    unsigned short* wcat_mn   = (unsigned short*)alloc((size_t)G4 * 1024 * 2);
    unsigned short* wcat_mc   = (unsigned short*)alloc((size_t)G4 * 1024 * 2);
    unsigned short* g1_bf     = (unsigned short*)alloc((size_t)D5 * G4 * 2);
    unsigned short* gg_bf     = (unsigned short*)alloc((size_t)D5 * D5 * 2);
    unsigned short* gq_bf     = (unsigned short*)alloc((size_t)D5 * D5 * 2);
    unsigned short* loc_bf    = (unsigned short*)alloc((size_t)D5 * D5 * 2);
    unsigned short* final_bf  = (unsigned short*)alloc((size_t)D5 * (3 * D5 + LDIM) * 2);
    unsigned short* ggin_bf   = (unsigned short*)alloc((size_t)BSZ * 49 * D5 * 2);
    unsigned short* locin_bf  = (unsigned short*)alloc((size_t)B2 * D5 * 2);
    float* locbn_s = (float*)alloc(D5 * 4);
    float* locbn_t = (float*)alloc(D5 * 4);
    unsigned short* xh_exp = (unsigned short*)alloc((size_t)BSZ * 1024 * 2);
    float* c_exp  = (float*)alloc((size_t)BSZ * D5 * 4);
    float* fexp   = (float*)alloc((size_t)BSZ * D5 * 4);
    unsigned short* fexp_bf = (unsigned short*)alloc((size_t)BSZ * D5 * 2);
    float* gg_out = (float*)alloc((size_t)BSZ * 49 * D5 * 4);
    float* gq_out = (float*)alloc((size_t)BSZ * D5 * 4);
    float* fglob  = (float*)alloc((size_t)BSZ * D5 * 4);
    float* flocal = (float*)alloc((size_t)B2 * D5 * 4);
    float* floc   = (float*)alloc((size_t)B2 * LDIM * 4);
    unsigned short* xh_fact = (unsigned short*)alloc((size_t)NFACT * 1024 * 2);
    float* gates  = (float*)alloc((size_t)NFACT * G4 * 4);
    float* c_fact = (float*)alloc((size_t)NFACT * D5 * 4);
    float* ffacts = (float*)alloc((size_t)NFACT * D5 * 4);
    unsigned short* ffacts_bf = (unsigned short*)alloc((size_t)NFACT * D5 * 2);
    unsigned short* z_bf = (unsigned short*)alloc((size_t)NFACT * G4 * 2);
    float* zh   = (float*)alloc((size_t)NFACT * D5 * 4);
    float* Zsc  = (float*)alloc((size_t)NFACT * 4);
    float* wbuf = (float*)alloc((size_t)B2 * FLEN * 4);
    float* f_m  = (float*)alloc((size_t)B2 * D5 * 4);
    float* mmem = (float*)alloc((size_t)B2 * D5 * 4);
    unsigned short* xh_ep = (unsigned short*)alloc((size_t)B2 * 1024 * 2);
    float* c_ep = (float*)alloc((size_t)B2 * D5 * 4);
    float* hpre = (float*)alloc((size_t)B2 * D5 * 4);
    float* eh   = (float*)alloc((size_t)B2 * D5 * 4);
    float* ec   = (float*)alloc((size_t)B2 * D5 * 4);
    unsigned short* feat_bf = (unsigned short*)alloc((size_t)B2 * (3 * D5 + LDIM) * 2);
    float* fvec = (float*)alloc((size_t)B2 * D5 * 4);

    auto cvt = [&](const float* s, unsigned short* d, int n) {
        k_cvt_bf16<<<(n + 255) / 256, 256, 0, stream>>>(s, d, n);
    };
    auto cat2 = [&](const float* a, const float* b, unsigned short* d) {
        k_cat2_bf16<<<(G4 * 1024 + 255) / 256, 256, 0, stream>>>(a, b, d, G4, D5, D5);
    };
    auto gemm = [&](const unsigned short* A, const unsigned short* B, const float* bias,
                    const float* scale, const float* shift, float* C,
                    int M, int N, int K, int lda, int ldb, int ldc, int act) {
        dim3 grid(N / 64, (M + 127) / 128);
        gemm_bf16_wmma<<<grid, 128, 0, stream>>>(A, B, bias, scale, shift, C,
                                                 M, N, K, lda, ldb, ldc, act);
    };
    auto zero = [&](void* p, size_t bytes) { hipMemsetAsync(p, 0, bytes, stream); };

    // ---- phase 0: weight prep --------------------------------------------
    cvt(emb, emb_bf, 10000 * D5);
    cat2(exp_Wih, exp_Whh, wcat_exp);
    cat2(fact_Wih, fact_Whh, wcat_fact);
    cat2(mn_Wih, mn_Whh, wcat_mn);
    cat2(mc_Wih, mc_Whh, wcat_mc);
    cvt(g1_W, g1_bf, D5 * G4);
    cvt(gg_W, gg_bf, D5 * D5);
    cvt(gq_W, gq_bf, D5 * D5);
    cvt(loc_W, loc_bf, D5 * D5);
    cvt(final_W, final_bf, D5 * (3 * D5 + LDIM));
    cvt(f_global_feat, ggin_bf, BSZ * 49 * D5);
    cvt(f_local_feat, locin_bf, B2 * D5);
    k_bn_eff<<<2, 256, 0, stream>>>(locbn_g, locbn_b, locbn_m, locbn_v, locbn_s, locbn_t, D5);

    // ---- phase 1: expression LSTM ----------------------------------------
    zero(xh_exp, (size_t)BSZ * 1024 * 2);
    zero(c_exp, (size_t)BSZ * D5 * 4);
    zero(fexp, (size_t)BSZ * D5 * 4);
    for (int t = 0; t < TLEN; ++t) {
        k_embed<<<BSZ, 128, 0, stream>>>(expression, TLEN, t, emb_bf, xh_exp, 1024);
        gemm(xh_exp, wcat_exp, nullptr, nullptr, nullptr, gates,
             BSZ, G4, 1024, 1024, 1024, G4, 0);
        k_cell_seq<<<(BSZ * D5 + 255) / 256, 256, 0, stream>>>(
            gates, exp_b, c_exp, xh_exp, fexp, e_mask, TLEN, t, BSZ);
    }
    k_bn<<<(BSZ * D5 + 255) / 256, 256, 0, stream>>>(
        fexp, fexp, fexp_bf, exp_bn_g, exp_bn_b, exp_bn_m, exp_bn_v, BSZ, D5);

    // ---- phase 2: global attention ---------------------------------------
    gemm(ggin_bf, gg_bf, gg_b, nullptr, nullptr, gg_out, BSZ * 49, D5, D5, D5, D5, D5, 0);
    gemm(fexp_bf, gq_bf, gq_b, nullptr, nullptr, gq_out, BSZ, D5, D5, D5, D5, D5, 0);
    k_att_glob<<<BSZ, 512, 0, stream>>>(gg_out, gq_out, gfc_W, gfc_b, f_global_feat, fglob);

    // ---- phase 3: local + position features ------------------------------
    gemm(locin_bf, loc_bf, loc_b, locbn_s, locbn_t, flocal, B2, D5, D5, D5, D5, D5, 2);
    k_pos<<<B2, LDIM, 0, stream>>>(locations, pos_W, pos_b,
                                   posbn_g, posbn_b, posbn_m, posbn_v, floc);

    // ---- phase 4: fact LSTM (dominant GEMMs) -----------------------------
    zero(xh_fact, (size_t)NFACT * 1024 * 2);
    zero(c_fact, (size_t)NFACT * D5 * 4);
    zero(ffacts, (size_t)NFACT * D5 * 4);
    for (int t = 0; t < TLEN; ++t) {
        k_embed<<<NFACT, 128, 0, stream>>>(facts, TLEN, t, emb_bf, xh_fact, 1024);
        gemm(xh_fact, wcat_fact, nullptr, nullptr, nullptr, gates,
             NFACT, G4, 1024, 1024, 1024, G4, 0);
        k_cell_seq<<<(NFACT * D5 + 255) / 256, 256, 0, stream>>>(
            gates, fact_b, c_fact, xh_fact, ffacts, maskp, TLEN, t, NFACT);
    }
    k_bn<<<(NFACT * D5 + 255) / 256, 256, 0, stream>>>(
        ffacts, ffacts, ffacts_bf, fact_bn_g, fact_bn_b, fact_bn_m, fact_bn_v, NFACT, D5);

    // ---- phase 5: init memory --------------------------------------------
    k_bcast_fm<<<(B2 * D5 + 255) / 256, 256, 0, stream>>>(fexp, f_m, mmem);

    // ---- phase 6: episodic memory x3 -------------------------------------
    for (int ep = 0; ep < 3; ++ep) {
        k_build_z<<<(NFACT * D5 + 255) / 256, 256, 0, stream>>>(ffacts, f_m, mmem, z_bf);
        gemm(z_bf, g1_bf, g1_b, nullptr, nullptr, zh, NFACT, D5, G4, G4, G4, D5, 1);
        k_g2<<<(NFACT + 7) / 8, 256, 0, stream>>>(zh, g2_W, g2_b, f_mask, Zsc, NFACT);
        k_softmax50<<<B2, 32, 0, stream>>>(Zsc, wbuf);
        zero(xh_ep, (size_t)B2 * 1024 * 2);
        zero(c_ep, (size_t)B2 * D5 * 4);
        zero(hpre, (size_t)B2 * D5 * 4);
        zero(eh, (size_t)B2 * D5 * 4);
        zero(ec, (size_t)B2 * D5 * 4);
        for (int s = 0; s < FLEN; ++s) {
            k_gather_fact<<<B2, 128, 0, stream>>>(ffacts_bf, xh_ep, s);
            gemm(xh_ep, wcat_mn, nullptr, nullptr, nullptr, gates,
                 B2, G4, 1024, 1024, 1024, G4, 0);
            k_cell_epi<<<(B2 * D5 + 255) / 256, 256, 0, stream>>>(
                gates, mn_b, c_ep, hpre, xh_ep, wbuf, s, ff_mask, eh, ec);
        }
        k_pack_eh_m<<<(B2 * D5 + 255) / 256, 256, 0, stream>>>(eh, mmem, xh_ep);
        gemm(xh_ep, wcat_mc, nullptr, nullptr, nullptr, gates,
             B2, G4, 1024, 1024, 1024, G4, 0);
        k_cell_mem<<<(B2 * D5 + 255) / 256, 256, 0, stream>>>(gates, mc_b, ec, mmem);
    }

    // ---- phase 7: final projection + scores ------------------------------
    k_pack_feat<<<B2, 256, 0, stream>>>(mmem, fglob, flocal, floc, feat_bf);
    gemm(feat_bf, final_bf, final_b, nullptr, nullptr, fvec,
         B2, D5, 3 * D5 + LDIM, 3 * D5 + LDIM, 3 * D5 + LDIM, D5, 0);
    k_scores<<<BSZ, 320, 0, stream>>>(fexp, fvec, c_mask, out);
}
